// NTMCell_56770877718771
// MI455X (gfx1250) — matrix operand order repaired and checked by reading, not compiled, not run on hardware
//
#include <hip/hip_runtime.h>
#include <hip/hip_bf16.h>

// ---------------------------------------------------------------------------
// NTM cell step for MI455X (gfx1250, wave32, WMMA).
// Memory-bound workload (~240 MB traffic, ~25 GFLOP): GEMMs use
// v_wmma_f32_16x16x32_bf16 (f32->bf16 convert on load, f32 accumulate).
// Block tile 128x32, double-buffered LDS W-tile in column-major (stride 40
// bf16 -> conflict-free ds_load_b128 B-fragments), hoisted address math.
// ---------------------------------------------------------------------------

constexpr int BATCH = 512;
constexpr int INPUT = 512;
constexpr int CDIM  = 1024;
constexpr int NMEM  = 256;
constexpr int DDIM  = 128;
constexpr int HEADS = 8;

constexpr int BSTRIDE = 40;               // bf16 per LDS column (padded, 80 B)
constexpr int BBUF    = 32 * BSTRIDE;     // one buffer: 32 cols

typedef __attribute__((ext_vector_type(16))) __bf16 v16bf;
typedef __attribute__((ext_vector_type(8)))  __bf16 v8bf;
typedef __attribute__((ext_vector_type(2)))  __bf16 bf16x2;
typedef __attribute__((ext_vector_type(8)))  float  v8f;

__device__ __forceinline__ float sigmoidf_(float x) { return 1.f / (1.f + __expf(-x)); }
__device__ __forceinline__ float softplusf_(float x) { return (x > 20.f) ? x : log1pf(__expf(x)); }

// Stage one 32k x 32n W^T tile into LDS (column-major, bf16). Each of the 256
// threads owns (col nc, even k kk2) and covers cols nc and nc+16:
// two coalesced global_load_b64 + two v_cvt_pk_bf16_f32 + two ds_store_b32.
__device__ __forceinline__ void stage_tile(
    const float* __restrict__ wpA, const float* __restrict__ wpB, int k0,
    __bf16* __restrict__ Bn, int nc, int kk2)
{
  float2 a = *(const float2*)(wpA + k0);
  float2 b = *(const float2*)(wpB + k0);
  bf16x2 pa = { (__bf16)a.x, (__bf16)a.y };
  bf16x2 pb = { (__bf16)b.x, (__bf16)b.y };
  *(bf16x2*)&Bn[nc * BSTRIDE + kk2]        = pa;
  *(bf16x2*)&Bn[(nc + 16) * BSTRIDE + kk2] = pb;
}

// ---------------------------------------------------------------------------
// One accumulate pass: acc{0,1} += A(M x K) @ W(N x K)^T over K, for one
// 128x32 block tile (this wave's rows = mrow..mrow+15, cols col0..col0+31).
// ---------------------------------------------------------------------------
__device__ __forceinline__ void gemm_pair(
    const float* __restrict__ A, const float* __restrict__ W, int K,
    int mrow, int col0, int half, int lane15, __bf16* __restrict__ Bt,
    v8f& acc0, v8f& acc1)
{
  const int nc  = threadIdx.x >> 4;         // 0..15
  const int kk2 = (threadIdx.x & 15) * 2;   // even k within tile
  const float* wpA = W + (size_t)(col0 + nc) * K + kk2;
  const float* wpB = W + (size_t)(col0 + nc + 16) * K + kk2;
  const float* ap  = A + (size_t)mrow * K;

  // prologue: stage k-tile 0 into buffer 0
  stage_tile(wpA, wpB, 0, Bt, nc, kk2);

  int cur = 0;
  for (int k0 = 0; k0 < K; k0 += 32) {
    __syncthreads();
    const int knext = k0 + 32;
    if (knext < K) {
      stage_tile(wpA, wpB, knext, Bt + (cur ^ 1) * BBUF, nc, kk2);
      // speculative prefetch of the k-tile after next (global_prefetch path)
      __builtin_prefetch(wpA + knext + 32, 0, 1);
      __builtin_prefetch(wpB + knext + 32, 0, 1);
    }

    // A 16x32 bf16 fragment (ISA 7.12.2): this lane needs two contiguous
    // 8-float runs: k0+8*half.. and k0+16+8*half.. -> 4x global_load_b128.
    const float4* pa0 = (const float4*)(ap + k0 + 8 * half);
    const float4* pa1 = (const float4*)(ap + k0 + 16 + 8 * half);
    float4 f0 = pa0[0], f1 = pa0[1];
    float4 f2 = pa1[0], f3 = pa1[1];
    v16bf a;
    a[0]  = (__bf16)f0.x; a[1]  = (__bf16)f0.y; a[2]  = (__bf16)f0.z; a[3]  = (__bf16)f0.w;
    a[4]  = (__bf16)f1.x; a[5]  = (__bf16)f1.y; a[6]  = (__bf16)f1.z; a[7]  = (__bf16)f1.w;
    a[8]  = (__bf16)f2.x; a[9]  = (__bf16)f2.y; a[10] = (__bf16)f2.z; a[11] = (__bf16)f2.w;
    a[12] = (__bf16)f3.x; a[13] = (__bf16)f3.y; a[14] = (__bf16)f3.z; a[15] = (__bf16)f3.w;

    // B 32x16 fragments: lane's 16 values = col (lane15 [+16]),
    // k = 8*half..8*half+7 and 16+8*half..16+8*half+7 (matches A) in column-major LDS:
    // two conflict-free ds_load_b128 per fragment.
    const __bf16* Bc = Bt + cur * BBUF;
    const __bf16* c0 = &Bc[lane15 * BSTRIDE];
    const __bf16* c1 = &Bc[(lane15 + 16) * BSTRIDE];
    v8bf b0l = *(const v8bf*)(c0 + 8 * half);
    v8bf b0h = *(const v8bf*)(c0 + 16 + 8 * half);
    v8bf b1l = *(const v8bf*)(c1 + 8 * half);
    v8bf b1h = *(const v8bf*)(c1 + 16 + 8 * half);
    v16bf b0 = __builtin_shufflevector(b0l, b0h, 0,1,2,3,4,5,6,7,8,9,10,11,12,13,14,15);
    v16bf b1 = __builtin_shufflevector(b1l, b1h, 0,1,2,3,4,5,6,7,8,9,10,11,12,13,14,15);

    acc0 = __builtin_amdgcn_wmma_f32_16x16x32_bf16(false, a, false, b0, (short)0, acc0, false, false);
    acc1 = __builtin_amdgcn_wmma_f32_16x16x32_bf16(false, a, false, b1, (short)0, acc1, false, false);
    cur ^= 1;
  }
  __syncthreads();   // Bt is reused by the caller's next pass
}

// ---------------------------------------------------------------------------
// Out[m,n] = sum_k A1[m,k] W1[n,k] (+ sum_k A2[m,k] W2[n,k]) + b1[n] (+ b2[n])
// Block = 256 threads = 8 waves; block tile = 128 rows x 32 cols; each wave
// owns two 16x16 output tiles. grid = (N/32, M/128); all dims exact multiples.
// ---------------------------------------------------------------------------
__global__ __launch_bounds__(256) void gemm2_wmma(
    const float* __restrict__ A1, const float* __restrict__ W1, int K1,
    const float* __restrict__ A2, const float* __restrict__ W2, int K2,
    const float* __restrict__ b1, const float* __restrict__ b2,
    float* __restrict__ Out, int Ncols)
{
  __shared__ __bf16 Bt[2 * BBUF];
  const int wave   = threadIdx.x >> 5;
  const int lane   = threadIdx.x & 31;
  const int lane15 = lane & 15;
  const int half   = lane >> 4;
  const int col0   = blockIdx.x * 32;
  const int row0   = blockIdx.y * 128 + wave * 16;
  const int mrow   = row0 + lane15;

  v8f acc0 = {}, acc1 = {};
  gemm_pair(A1, W1, K1, mrow, col0, half, lane15, Bt, acc0, acc1);
  if (A2) {
    gemm_pair(A2, W2, K2, mrow, col0, half, lane15, Bt, acc0, acc1);
  }

  // C/D 16x16 f32 layout: VGPR r -> row (r + 8*half), col = lane15.
  const int ncol0 = col0 + lane15;
  const int ncol1 = ncol0 + 16;
  float bias0 = (b1 ? b1[ncol0] : 0.f) + (b2 ? b2[ncol0] : 0.f);
  float bias1 = (b1 ? b1[ncol1] : 0.f) + (b2 ? b2[ncol1] : 0.f);
  const int rb = row0 + half * 8;
#pragma unroll
  for (int rr = 0; rr < 8; ++rr) {
    Out[(size_t)(rb + rr) * Ncols + ncol0] = acc0[rr] + bias0;
    Out[(size_t)(rb + rr) * Ncols + ncol1] = acc1[rr] + bias1;
  }
}

// ---------------------------------------------------------------------------
// inp = concat(x, r_prev.flatten())  -> (B, INPUT + H*D) = (512, 1536)
// ---------------------------------------------------------------------------
__global__ void pack_inp(const float* __restrict__ x, const float* __restrict__ rp,
                         float* __restrict__ inp)
{
  int idx = blockIdx.x * blockDim.x + threadIdx.x;   // over B*1536
  int b = idx / (INPUT + HEADS * DDIM);
  int c = idx % (INPUT + HEADS * DDIM);
  inp[idx] = (c < INPUT) ? x[(size_t)b * INPUT + c]
                         : rp[(size_t)b * (HEADS * DDIM) + (c - INPUT)];
}

// ---------------------------------------------------------------------------
// LSTM pointwise: gates g (B x 4C, order i,f,g,o) + c_prev -> h_new, c_new
// ---------------------------------------------------------------------------
__global__ void lstm_pw(const float* __restrict__ g, const float* __restrict__ c_prev,
                        float* __restrict__ h_out, float* __restrict__ c_out)
{
  int idx = blockIdx.x * blockDim.x + threadIdx.x;   // over B*C
  int b = idx / CDIM;
  int c = idx % CDIM;
  const float* gr = g + (size_t)b * 4 * CDIM;
  float gi = sigmoidf_(gr[c]);
  float gf = sigmoidf_(gr[CDIM + c]);
  float gg = tanhf(gr[2 * CDIM + c]);
  float go = sigmoidf_(gr[3 * CDIM + c]);
  float cn = gf * c_prev[idx] + gi * gg;
  c_out[idx] = cn;
  h_out[idx] = go * tanhf(cn);
}

// erase = sigmoid(erase_pre), addv = tanh(addv_pre)  (in place)
__global__ void act_er_ad(float* __restrict__ er, float* __restrict__ ad)
{
  int idx = blockIdx.x * blockDim.x + threadIdx.x;   // over B*H*D
  er[idx] = sigmoidf_(er[idx]);
  ad[idx] = tanhf(ad[idx]);
}

// ---------------------------------------------------------------------------
// Small heads: out[b,o] = dot(A[b,:], W[o,:]) + bias[o], K = CDIM.
// grid = (B, Nout), block = 256 with LDS tree reduction.
// ---------------------------------------------------------------------------
__global__ void gemv_small(const float* __restrict__ A, const float* __restrict__ W,
                           const float* __restrict__ bias, float* __restrict__ out,
                           int Nout)
{
  __shared__ float red[256];
  int b = blockIdx.x, o = blockIdx.y, t = threadIdx.x;
  const float* ap = A + (size_t)b * CDIM;
  const float* wp = W + (size_t)o * CDIM;
  float acc = 0.f;
  for (int k = t; k < CDIM; k += 256) acc += ap[k] * wp[k];
  red[t] = acc;
  __syncthreads();
  for (int s = 128; s > 0; s >>= 1) {
    if (t < s) red[t] += red[t + s];
    __syncthreads();
  }
  if (t == 0) out[(size_t)b * Nout + o] = red[0] + bias[o];
}

// ---------------------------------------------------------------------------
// Content addressing: w[b,h,:] = softmax_n( softplus(beta_raw[b,h]) *
//   cos(M[b,n,:], K[b,h,:]) ) with per-vector normalization (eps 1e-12).
// grid = (B, H), block = NMEM (256) threads (one per memory row).
// ---------------------------------------------------------------------------
__global__ __launch_bounds__(NMEM) void content_w(
    const float* __restrict__ M, const float* __restrict__ Kp,
    const float* __restrict__ beta_raw, float* __restrict__ w_out)
{
  __shared__ float sK[DDIM];
  __shared__ float red[NMEM];
  int b = blockIdx.x, h = blockIdx.y, n = threadIdx.x;

  const float* kp = Kp + ((size_t)b * HEADS + h) * DDIM;
  if (n < DDIM) sK[n] = kp[n];
  __syncthreads();

  float kn = 0.f;
#pragma unroll 8
  for (int d = 0; d < DDIM; ++d) kn += sK[d] * sK[d];
  kn = fmaxf(sqrtf(kn), 1e-12f);

  const float* mr = M + ((size_t)b * NMEM + n) * DDIM;
  float dot = 0.f, mn = 0.f;
#pragma unroll 8
  for (int d = 0; d < DDIM; ++d) {
    float mv = mr[d];
    dot += mv * sK[d];
    mn += mv * mv;
  }
  mn = fmaxf(sqrtf(mn), 1e-12f);

  float beta = softplusf_(beta_raw[(size_t)b * HEADS + h]);
  float logit = beta * dot / (mn * kn);

  red[n] = logit;
  __syncthreads();
  for (int s = NMEM / 2; s > 0; s >>= 1) {
    if (n < s) red[n] = fmaxf(red[n], red[n + s]);
    __syncthreads();
  }
  float mx = red[0];
  __syncthreads();
  float ex = __expf(logit - mx);
  red[n] = ex;
  __syncthreads();
  for (int s = NMEM / 2; s > 0; s >>= 1) {
    if (n < s) red[n] += red[n + s];
    __syncthreads();
  }
  w_out[((size_t)b * HEADS + h) * NMEM + n] = ex / red[0];
}

// ---------------------------------------------------------------------------
// M = M_prev * (1 - sum_h w*erase) + sum_h w*addv.  grid=(B,N), block=D.
// ---------------------------------------------------------------------------
__global__ __launch_bounds__(DDIM) void mem_update(
    const float* __restrict__ Mp, const float* __restrict__ ww,
    const float* __restrict__ er, const float* __restrict__ ad,
    float* __restrict__ Mout)
{
  int b = blockIdx.x, n = blockIdx.y, d = threadIdx.x;
  float es = 0.f, as = 0.f;
#pragma unroll
  for (int h = 0; h < HEADS; ++h) {
    float wv = ww[((size_t)b * HEADS + h) * NMEM + n];
    es += wv * er[((size_t)b * HEADS + h) * DDIM + d];
    as += wv * ad[((size_t)b * HEADS + h) * DDIM + d];
  }
  size_t idx = ((size_t)b * NMEM + n) * DDIM + d;
  Mout[idx] = Mp[idx] * (1.f - es) + as;
}

// ---------------------------------------------------------------------------
// Circular shift (softmax of 3 shift logits) + sharpen (gamma) + renormalize.
// grid=(B,H), block=NMEM.
// ---------------------------------------------------------------------------
__global__ __launch_bounds__(NMEM) void shift_sharpen(
    const float* __restrict__ wc, const float* __restrict__ sh_raw,
    const float* __restrict__ ga_raw, float* __restrict__ wr_out)
{
  __shared__ float s[3];
  __shared__ float buf[NMEM];
  __shared__ float red[NMEM];
  int b = blockIdx.x, h = blockIdx.y, n = threadIdx.x;

  if (n == 0) {
    const float* shp = sh_raw + ((size_t)b * HEADS + h) * 3;
    float a0 = shp[0], a1 = shp[1], a2 = shp[2];
    float mx = fmaxf(a0, fmaxf(a1, a2));
    float e0 = __expf(a0 - mx), e1 = __expf(a1 - mx), e2 = __expf(a2 - mx);
    float sum = e0 + e1 + e2;
    s[0] = e0 / sum; s[1] = e1 / sum; s[2] = e2 / sum;
  }
  buf[n] = wc[((size_t)b * HEADS + h) * NMEM + n];
  __syncthreads();

  // w_s[n] = w[n+1]*s0 + w[n]*s1 + w[n-1]*s2   (circular)
  float ws = buf[(n + 1) & (NMEM - 1)] * s[0] + buf[n] * s[1] +
             buf[(n + NMEM - 1) & (NMEM - 1)] * s[2];
  float g = 1.f + softplusf_(ga_raw[(size_t)b * HEADS + h]);
  float wg = powf(fmaxf(ws, 0.f), g);

  red[n] = wg;
  __syncthreads();
  for (int st = NMEM / 2; st > 0; st >>= 1) {
    if (n < st) red[n] += red[n + st];
    __syncthreads();
  }
  wr_out[((size_t)b * HEADS + h) * NMEM + n] = wg / (red[0] + 1e-12f);
}

// ---------------------------------------------------------------------------
// r[b,h,d] = sum_n w_r[b,h,n] * M[b,n,d].  grid=(B,H), block=D.
// ---------------------------------------------------------------------------
__global__ __launch_bounds__(DDIM) void read_mem(
    const float* __restrict__ wr, const float* __restrict__ M,
    float* __restrict__ r)
{
  int b = blockIdx.x, h = blockIdx.y, d = threadIdx.x;
  const float* wp = wr + ((size_t)b * HEADS + h) * NMEM;
  const float* mp = M + (size_t)b * NMEM * DDIM + d;
  float acc = 0.f;
#pragma unroll 4
  for (int n = 0; n < NMEM; ++n) acc += wp[n] * mp[(size_t)n * DDIM];
  r[((size_t)b * HEADS + h) * DDIM + d] = acc;
}

// ---------------------------------------------------------------------------
extern "C" void kernel_launch(void* const* d_in, const int* in_sizes, int n_in,
                              void* d_out, int out_size, void* d_ws, size_t ws_size,
                              hipStream_t stream)
{
  (void)in_sizes; (void)n_in; (void)out_size; (void)ws_size;

  const float* x      = (const float*)d_in[0];
  const float* h1     = (const float*)d_in[1];
  const float* c1     = (const float*)d_in[2];
  const float* h2     = (const float*)d_in[3];
  const float* c2     = (const float*)d_in[4];
  /* d_in[5] = w_prev, unused by the math */
  const float* r_prev = (const float*)d_in[6];
  const float* M_prev = (const float*)d_in[7];
  const float* W_ih1  = (const float*)d_in[8];
  const float* W_hh1  = (const float*)d_in[9];
  const float* b_ih1  = (const float*)d_in[10];
  const float* b_hh1  = (const float*)d_in[11];
  const float* W_ih2  = (const float*)d_in[12];
  const float* W_hh2  = (const float*)d_in[13];
  const float* b_ih2  = (const float*)d_in[14];
  const float* b_hh2  = (const float*)d_in[15];
  const float* W_rk   = (const float*)d_in[16]; const float* b_rk = (const float*)d_in[17];
  const float* W_wk   = (const float*)d_in[18]; const float* b_wk = (const float*)d_in[19];
  const float* W_rb   = (const float*)d_in[20]; const float* b_rb = (const float*)d_in[21];
  const float* W_wb   = (const float*)d_in[22]; const float* b_wb = (const float*)d_in[23];
  const float* W_er   = (const float*)d_in[24]; const float* b_er = (const float*)d_in[25];
  const float* W_ad   = (const float*)d_in[26]; const float* b_ad = (const float*)d_in[27];
  const float* W_sh   = (const float*)d_in[28]; const float* b_sh = (const float*)d_in[29];
  const float* W_ga   = (const float*)d_in[30]; const float* b_ga = (const float*)d_in[31];

  // d_out layout (reference tuple order, flat f32)
  float* out = (float*)d_out;
  float* h1n = out;
  float* c1n = out + 1 * (size_t)BATCH * CDIM;
  float* h2n = out + 2 * (size_t)BATCH * CDIM;
  float* c2n = out + 3 * (size_t)BATCH * CDIM;
  float* w_r = out + 4 * (size_t)BATCH * CDIM;
  float* r_o = w_r + (size_t)BATCH * HEADS * NMEM;
  float* M_o = r_o + (size_t)BATCH * HEADS * DDIM;

  // workspace
  float* w = (float*)d_ws;
  size_t off = 0;
  float* inp = w + off; off += (size_t)BATCH * (INPUT + HEADS * DDIM); // 512x1536
  float* g   = w + off; off += (size_t)BATCH * 4 * CDIM;               // gates (reused)
  float* RK  = w + off; off += (size_t)BATCH * CDIM;
  float* WK  = w + off; off += (size_t)BATCH * CDIM;
  float* ER  = w + off; off += (size_t)BATCH * CDIM;
  float* AD  = w + off; off += (size_t)BATCH * CDIM;
  float* Brw = w + off; off += (size_t)BATCH * HEADS;
  float* Bww = w + off; off += (size_t)BATCH * HEADS;
  float* SHw = w + off; off += (size_t)BATCH * HEADS * 3;
  float* GAw = w + off; off += (size_t)BATCH * HEADS;
  float* ww_ = w + off; off += (size_t)BATCH * HEADS * NMEM;
  float* wc_ = w + off; off += (size_t)BATCH * HEADS * NMEM;

  const int KIN = INPUT + HEADS * DDIM;  // 1536

  // 1) concat input
  pack_inp<<<(BATCH * KIN) / 256, 256, 0, stream>>>(x, r_prev, inp);

  // 2) LSTM1: g = inp@W_ih1^T + h1@W_hh1^T + biases  (fused dual-pair WMMA GEMM)
  gemm2_wmma<<<dim3(4 * CDIM / 32, BATCH / 128), 256, 0, stream>>>(
      inp, W_ih1, KIN, h1, W_hh1, CDIM, b_ih1, b_hh1, g, 4 * CDIM);
  lstm_pw<<<(BATCH * CDIM) / 256, 256, 0, stream>>>(g, c1, h1n, c1n);

  // 3) LSTM2
  gemm2_wmma<<<dim3(4 * CDIM / 32, BATCH / 128), 256, 0, stream>>>(
      h1n, W_ih2, CDIM, h2, W_hh2, CDIM, b_ih2, b_hh2, g, 4 * CDIM);
  lstm_pw<<<(BATCH * CDIM) / 256, 256, 0, stream>>>(g, c2, h2n, c2n);

  // 4) wide head projections (WMMA)
  gemm2_wmma<<<dim3(CDIM / 32, BATCH / 128), 256, 0, stream>>>(
      h2n, W_rk, CDIM, nullptr, nullptr, 0, b_rk, nullptr, RK, CDIM);
  gemm2_wmma<<<dim3(CDIM / 32, BATCH / 128), 256, 0, stream>>>(
      h2n, W_wk, CDIM, nullptr, nullptr, 0, b_wk, nullptr, WK, CDIM);
  gemm2_wmma<<<dim3(CDIM / 32, BATCH / 128), 256, 0, stream>>>(
      h2n, W_er, CDIM, nullptr, nullptr, 0, b_er, nullptr, ER, CDIM);
  gemm2_wmma<<<dim3(CDIM / 32, BATCH / 128), 256, 0, stream>>>(
      h2n, W_ad, CDIM, nullptr, nullptr, 0, b_ad, nullptr, AD, CDIM);
  act_er_ad<<<(BATCH * CDIM) / 256, 256, 0, stream>>>(ER, AD);

  // 5) small head projections (beta_r, beta_w, shifts, gamma) — raw, activated at use
  gemv_small<<<dim3(BATCH, HEADS), 256, 0, stream>>>(h2n, W_rb, b_rb, Brw, HEADS);
  gemv_small<<<dim3(BATCH, HEADS), 256, 0, stream>>>(h2n, W_wb, b_wb, Bww, HEADS);
  gemv_small<<<dim3(BATCH, HEADS * 3), 256, 0, stream>>>(h2n, W_sh, b_sh, SHw, HEADS * 3);
  gemv_small<<<dim3(BATCH, HEADS), 256, 0, stream>>>(h2n, W_ga, b_ga, GAw, HEADS);

  // 6) write addressing + memory update
  content_w<<<dim3(BATCH, HEADS), NMEM, 0, stream>>>(M_prev, WK, Bww, ww_);
  mem_update<<<dim3(BATCH, NMEM), DDIM, 0, stream>>>(M_prev, ww_, ER, AD, M_o);

  // 7) read addressing: content -> shift -> sharpen -> read
  content_w<<<dim3(BATCH, HEADS), NMEM, 0, stream>>>(M_o, RK, Brw, wc_);
  shift_sharpen<<<dim3(BATCH, HEADS), NMEM, 0, stream>>>(wc_, SHw, GAw, w_r);
  read_mem<<<dim3(BATCH, HEADS), DDIM, 0, stream>>>(w_r, M_o, r_o);
}
